// TemporalCMFuser_83605833384473
// MI455X (gfx1250) — matrix operand order repaired and checked by reading, not compiled, not run on hardware
//
#include <hip/hip_runtime.h>
#include <hip/hip_bf16.h>

// ---------------- problem constants ----------------
#define DIMC   1024
#define TT     32
#define BB     32
#define NMOD   3
#define SS     96          // NMOD * TT
#define NHEAD  16
#define HD     64
#define MLPC   4096
#define NDEPTH 4
#define ROWS   (BB * SS)   // 3072

typedef __attribute__((ext_vector_type(16))) _Float16 v16h;
typedef __attribute__((ext_vector_type(8)))  _Float16 v8h;
typedef __attribute__((ext_vector_type(8)))  float    v8f;

__device__ __forceinline__ void wait_asynccnt0() {
#if defined(__has_builtin)
#if __has_builtin(__builtin_amdgcn_s_wait_asynccnt)
  __builtin_amdgcn_s_wait_asynccnt(0);
  return;
#else
  asm volatile("s_wait_asynccnt 0x0" ::: "memory");
  return;
#endif
#else
  asm volatile("s_wait_asynccnt 0x0" ::: "memory");
#endif
}

// per-lane async copy of 16 bytes global -> LDS (gfx1250 ASYNCcnt path)
__device__ __forceinline__ void async_copy_b128(const void* gptr, void* lptr) {
  unsigned lds = (unsigned)(size_t)lptr;               // low 32 bits = LDS offset
  unsigned long long ga = (unsigned long long)(size_t)gptr;
  asm volatile("global_load_async_to_lds_b128 %0, %1, off"
               :: "v"(lds), "v"(ga) : "memory");
}

// ---------------- embeddings ----------------
__global__ __launch_bounds__(256) void embed_kernel(
    const float* __restrict__ f0, const float* __restrict__ f1,
    const float* __restrict__ f2, const float* __restrict__ pos,
    const float* __restrict__ mod, float* __restrict__ x) {
  int idx = blockIdx.x * 256 + threadIdx.x;      // ROWS*DIMC total
  int c = idx % DIMC;
  int s = (idx / DIMC) % SS;
  int b = idx / (DIMC * SS);
  int m = s / TT, t = s % TT;
  const float* f = (m == 0) ? f0 : (m == 1) ? f1 : f2;
  x[idx] = f[((size_t)b * TT + t) * DIMC + c] + pos[t * DIMC + c] + mod[m * DIMC + c];
}

// ---------------- layernorm (one block per row of 1024) ----------------
template <typename OT>
__global__ __launch_bounds__(256) void ln_kernel(
    const float* __restrict__ X, const float* __restrict__ w,
    const float* __restrict__ bb, OT* __restrict__ out) {
  __shared__ float red[8], red2[8];
  __shared__ float smean, sinv;
  const int row = blockIdx.x;
  const int tid = threadIdx.x;
  const float* xr = X + (size_t)row * DIMC;
  float v0[4];
  float s = 0.f, s2 = 0.f;
#pragma unroll
  for (int j = 0; j < 4; ++j) {
    float v = xr[tid + j * 256];
    v0[j] = v; s += v; s2 += v * v;
  }
#pragma unroll
  for (int off = 16; off > 0; off >>= 1) {
    s  += __shfl_xor(s, off);
    s2 += __shfl_xor(s2, off);
  }
  if ((tid & 31) == 0) { red[tid >> 5] = s; red2[tid >> 5] = s2; }
  __syncthreads();
  if (tid == 0) {
    float ts = 0.f, ts2 = 0.f;
    for (int i = 0; i < 8; ++i) { ts += red[i]; ts2 += red2[i]; }
    float m  = ts * (1.0f / DIMC);
    float var = ts2 * (1.0f / DIMC) - m * m;
    smean = m;
    sinv  = rsqrtf(var + 1e-6f);
  }
  __syncthreads();
  float m = smean, inv = sinv;
#pragma unroll
  for (int j = 0; j < 4; ++j) {
    int c = tid + j * 256;
    out[(size_t)row * DIMC + c] = (OT)((v0[j] - m) * inv * w[c] + bb[c]);
  }
}

// ---------------- f32 -> f16 weight convert ----------------
__global__ __launch_bounds__(256) void cvt_kernel(
    const float* __restrict__ src, _Float16* __restrict__ dst) {
  int idx = blockIdx.x * 256 + threadIdx.x;
  dst[idx] = (_Float16)src[idx];
}

// ---------------- WMMA f16 GEMM: C[M,N] = A[M,K] @ Bm[K,N] ----------------
// Block tile 128x128, 8 waves, each wave 32x64 (2x4 tiles of 16x16).
// Double-buffered LDS; A tile staged with global_load_async_to_lds_b128
// (ASYNCcnt), B tile software-pipelined through VGPRs (transpose to [n][k]).
#define BM 128
#define BN 128
#define BK 32
__global__ __launch_bounds__(256) void gemm_wmma(
    const _Float16* __restrict__ A, const _Float16* __restrict__ Bm,
    float* __restrict__ C, int M, int N, int K) {
  __shared__ __attribute__((aligned(16))) _Float16 As[2][BM * BK];  // [m][k]
  __shared__ __attribute__((aligned(16))) _Float16 Bs[2][BN * BK];  // [n][k]
  const int tid   = threadIdx.x;
  const int lane  = tid & 31;
  const int wave  = tid >> 5;
  const int waveM = wave & 3;    // 4 row groups * 32 rows
  const int waveN = wave >> 2;   // 2 col groups * 64 cols
  const int blockM = blockIdx.y * BM;
  const int blockN = blockIdx.x * BN;
  const int lr = lane & 15;
  const int lh = lane >> 4;

  // staging coordinates (512 x 16B chunks per tile, 2 per thread)
  const int ar0 = tid >> 1;                 // A chunk0: row,  col8
  const int ac0 = (tid & 1) * 8;
  const int ar1 = (tid + 256) >> 1;
  const int ac1 = ((tid + 256) & 1) * 8;
  const int bk0 = tid >> 4;                 // B chunk0: k, n8
  const int bn0 = (tid & 15) * 8;
  const int bk1 = (tid + 256) >> 4;
  const int bn1 = ((tid + 256) & 15) * 8;

  v8f acc[2][4];
#pragma unroll
  for (int mi = 0; mi < 2; ++mi)
#pragma unroll
    for (int ni = 0; ni < 4; ++ni)
#pragma unroll
      for (int j = 0; j < 8; ++j) acc[mi][ni][j] = 0.0f;

  // ---- preamble: stage tile 0 ----
  async_copy_b128(&A[(size_t)(blockM + ar0) * K + ac0], &As[0][ar0 * BK + ac0]);
  async_copy_b128(&A[(size_t)(blockM + ar1) * K + ac1], &As[0][ar1 * BK + ac1]);
  {
    v8h t0 = *(const v8h*)&Bm[(size_t)bk0 * N + blockN + bn0];
    v8h t1 = *(const v8h*)&Bm[(size_t)bk1 * N + blockN + bn1];
#pragma unroll
    for (int j = 0; j < 8; ++j) Bs[0][(bn0 + j) * BK + bk0] = t0[j];
#pragma unroll
    for (int j = 0; j < 8; ++j) Bs[0][(bn1 + j) * BK + bk1] = t1[j];
  }
  wait_asynccnt0();
  __syncthreads();

  for (int kb = 0; kb < K; kb += BK) {
    const int buf = (kb >> 5) & 1;
    const int nxt = buf ^ 1;
    const bool hasNext = (kb + BK) < K;

    // issue next tile's traffic before compute so it overlaps the WMMAs
    v8h tb0, tb1;
    if (hasNext) {
      async_copy_b128(&A[(size_t)(blockM + ar0) * K + kb + BK + ac0],
                      &As[nxt][ar0 * BK + ac0]);
      async_copy_b128(&A[(size_t)(blockM + ar1) * K + kb + BK + ac1],
                      &As[nxt][ar1 * BK + ac1]);
      tb0 = *(const v8h*)&Bm[(size_t)(kb + BK + bk0) * N + blockN + bn0];
      tb1 = *(const v8h*)&Bm[(size_t)(kb + BK + bk1) * N + blockN + bn1];
      if (kb + 2 * BK < K)  // look-ahead prefetch for the B stream
        __builtin_prefetch(&Bm[(size_t)(kb + 2 * BK + bk0) * N + blockN], 0, 1);
    }

    // ---- compute on buf ----
    // A frags: lane holds row (lr), K = lh*8 + j (j<8) and 16 + lh*8 + j
    v16h af[2];
#pragma unroll
    for (int mi = 0; mi < 2; ++mi) {
      const _Float16* ap = &As[buf][(waveM * 32 + mi * 16 + lr) * BK + lh * 8];
      v8h lo = *(const v8h*)ap;
      v8h hi = *(const v8h*)(ap + 16);
#pragma unroll
      for (int j = 0; j < 8; ++j) { af[mi][j] = lo[j]; af[mi][8 + j] = hi[j]; }
    }
    // B frags: lane holds col (lr), K = lh*8 + j (j<8) and 16 + lh*8 + j
    v16h bf[4];
#pragma unroll
    for (int ni = 0; ni < 4; ++ni) {
      const _Float16* bp = &Bs[buf][(waveN * 64 + ni * 16 + lr) * BK + lh * 8];
      v8h lo = *(const v8h*)bp;
      v8h hi = *(const v8h*)(bp + 16);
#pragma unroll
      for (int j = 0; j < 8; ++j) { bf[ni][j] = lo[j]; bf[ni][8 + j] = hi[j]; }
    }
#pragma unroll
    for (int mi = 0; mi < 2; ++mi)
#pragma unroll
      for (int ni = 0; ni < 4; ++ni)
        acc[mi][ni] = __builtin_amdgcn_wmma_f32_16x16x32_f16(
            false, af[mi], false, bf[ni], (short)0, acc[mi][ni], false, false);

    // ---- finish staging next tile (transpose B into LDS) ----
    if (hasNext) {
#pragma unroll
      for (int j = 0; j < 8; ++j) Bs[nxt][(bn0 + j) * BK + bk0] = tb0[j];
#pragma unroll
      for (int j = 0; j < 8; ++j) Bs[nxt][(bn1 + j) * BK + bk1] = tb1[j];
    }
    wait_asynccnt0();     // drain async A copies (ASYNCcnt)
    __syncthreads();      // emits s_wait_dscnt 0 + s_barrier for B stores
  }

  // C/D layout: VGPR j: lanes 0-15 -> M=j, lanes 16-31 -> M=j+8; N = lane&15
#pragma unroll
  for (int mi = 0; mi < 2; ++mi)
#pragma unroll
    for (int ni = 0; ni < 4; ++ni)
#pragma unroll
      for (int j = 0; j < 8; ++j) {
        int row = blockM + waveM * 32 + mi * 16 + lh * 8 + j;
        int col = blockN + waveN * 64 + ni * 16 + lr;
        C[(size_t)row * N + col] = acc[mi][ni][j];
      }
}

// ---------------- attention: one block per (b, head) ----------------
__global__ __launch_bounds__(256) void attn_kernel(
    const float* __restrict__ qkv, _Float16* __restrict__ o16,
    float* __restrict__ attn_out, int layer) {
  __shared__ _Float16 q16[SS * HD];
  __shared__ _Float16 k16[SS * HD];
  __shared__ float p[SS * SS];
  const int b = blockIdx.x / NHEAD;
  const int h = blockIdx.x % NHEAD;
  const int tid = threadIdx.x;
  const float scale = 0.125f;  // 64^-0.5

  for (int idx = tid; idx < SS * HD; idx += 256) {
    int s = idx / HD, d = idx % HD;
    const float* base = qkv + (size_t)(b * SS + s) * (3 * DIMC) + h * HD + d;
    q16[idx] = (_Float16)base[0];
    k16[idx] = (_Float16)base[DIMC];
  }
  __syncthreads();

  for (int idx = tid; idx < SS * SS; idx += 256) {
    int qi = idx / SS, kj = idx % SS;
    float sc;
    if ((kj & (TT - 1)) > (qi & (TT - 1))) {
      sc = -3.0e38f;  // block-tiled causal mask
    } else {
      float a = 0.f;
#pragma unroll 8
      for (int d = 0; d < HD; ++d)
        a += (float)q16[qi * HD + d] * (float)k16[kj * HD + d];
      sc = a * scale;
    }
    p[idx] = sc;
  }
  __syncthreads();

  if (tid < SS) {
    float* row = &p[tid * SS];
    float mx = row[0];
    for (int j = 1; j < SS; ++j) mx = fmaxf(mx, row[j]);
    float sum = 0.f;
    for (int j = 0; j < SS; ++j) { float e = __expf(row[j] - mx); row[j] = e; sum += e; }
    float inv = 1.f / sum;
    float* orow = attn_out +
        ((size_t)((b * NDEPTH + layer) * NHEAD + h) * SS + tid) * SS;
    for (int j = 0; j < SS; ++j) { float pv = row[j] * inv; row[j] = pv; orow[j] = pv; }
  }
  __syncthreads();

  for (int idx = tid; idx < SS * HD; idx += 256) {
    int qi = idx / HD, d = idx % HD;
    const float* vbase = qkv + (size_t)(b * SS) * (3 * DIMC) + 2 * DIMC + h * HD + d;
    const float* prow = &p[qi * SS];
    float a = 0.f;
    for (int j = 0; j < SS; ++j) a += prow[j] * vbase[(size_t)j * (3 * DIMC)];
    o16[(size_t)(b * SS + qi) * DIMC + h * HD + d] = (_Float16)a;
  }
}

// ---------------- residual add: x += c + bias ----------------
__global__ __launch_bounds__(256) void addres_kernel(
    float* __restrict__ x, const float* __restrict__ c,
    const float* __restrict__ bias, int N) {
  int idx = blockIdx.x * 256 + threadIdx.x;
  x[idx] += c[idx] + bias[idx % N];
}

// ---------------- exact GELU + bias, f16 out ----------------
__global__ __launch_bounds__(256) void gelu_kernel(
    const float* __restrict__ m, const float* __restrict__ bias,
    _Float16* __restrict__ out) {
  int idx = blockIdx.x * 256 + threadIdx.x;
  float v = m[idx] + bias[idx % MLPC];
  float g = 0.5f * v * (1.0f + erff(v * 0.70710678118654752f));
  out[idx] = (_Float16)g;
}

// ---------------- mean-pool over modalities ----------------
__global__ __launch_bounds__(256) void fuse_kernel(
    const float* __restrict__ xn, float* __restrict__ out) {
  int idx = blockIdx.x * 256 + threadIdx.x;  // B*T*DIM total
  int c = idx % DIMC;
  int bt = idx / DIMC;
  int t = bt % TT;
  int b = bt / TT;
  float s = 0.f;
#pragma unroll
  for (int m = 0; m < NMOD; ++m)
    s += xn[((size_t)(b * SS) + m * TT + t) * DIMC + c];
  out[idx] = s * (1.0f / 3.0f);
}

// ---------------- host-side orchestration ----------------
extern "C" void kernel_launch(void* const* d_in, const int* in_sizes, int n_in,
                              void* d_out, int out_size, void* d_ws, size_t ws_size,
                              hipStream_t stream) {
  (void)in_sizes; (void)n_in; (void)out_size; (void)ws_size;
  const float* feat0  = (const float*)d_in[0];
  const float* feat1  = (const float*)d_in[1];
  const float* feat2  = (const float*)d_in[2];
  const float* pos    = (const float*)d_in[3];
  const float* mode   = (const float*)d_in[4];
  const float* n1w    = (const float*)d_in[5];
  const float* n1b    = (const float*)d_in[6];
  const float* qkv_w  = (const float*)d_in[7];
  const float* proj_w = (const float*)d_in[8];
  const float* proj_b = (const float*)d_in[9];
  const float* n2w    = (const float*)d_in[10];
  const float* n2b    = (const float*)d_in[11];
  const float* fc1_w  = (const float*)d_in[12];
  const float* fc1_b  = (const float*)d_in[13];
  const float* fc2_w  = (const float*)d_in[14];
  const float* fc2_b  = (const float*)d_in[15];
  const float* fnw    = (const float*)d_in[16];
  const float* fnb    = (const float*)d_in[17];

  // workspace layout (all offsets 4KB-aligned)
  char* w = (char*)d_ws;
  float*    x    = (float*)(w);                      // ROWS*DIMC f32   (12.6 MB)
  float*    bigC = (float*)(w + 12582912);           // ROWS*MLPC f32   (50.3 MB)
  _Float16* m16  = (_Float16*)(w + 62914560);        // ROWS*MLPC f16   (25.2 MB)
  _Float16* h16  = (_Float16*)(w + 88080384);        // ROWS*DIMC f16   (6.3 MB)
  _Float16* o16  = (_Float16*)(w + 94371840);        // ROWS*DIMC f16   (6.3 MB)
  _Float16* w16  = (_Float16*)(w + 100663296);       // MLPC*DIMC f16   (8.4 MB)

  float* out_fused = (float*)d_out;
  float* out_attn  = (float*)d_out + (size_t)BB * TT * DIMC;

  dim3 blk(256);
  embed_kernel<<<(ROWS * DIMC) / 256, blk, 0, stream>>>(feat0, feat1, feat2, pos, mode, x);

  for (int l = 0; l < NDEPTH; ++l) {
    // ---- attention ----
    ln_kernel<_Float16><<<ROWS, blk, 0, stream>>>(x, n1w + l * DIMC, n1b + l * DIMC, h16);
    cvt_kernel<<<(DIMC * 3 * DIMC) / 256, blk, 0, stream>>>(qkv_w + (size_t)l * DIMC * 3 * DIMC, w16);
    gemm_wmma<<<dim3((3 * DIMC) / BN, ROWS / BM), blk, 0, stream>>>(h16, w16, bigC, ROWS, 3 * DIMC, DIMC);
    attn_kernel<<<BB * NHEAD, blk, 0, stream>>>(bigC, o16, out_attn, l);
    cvt_kernel<<<(DIMC * DIMC) / 256, blk, 0, stream>>>(proj_w + (size_t)l * DIMC * DIMC, w16);
    gemm_wmma<<<dim3(DIMC / BN, ROWS / BM), blk, 0, stream>>>(o16, w16, bigC, ROWS, DIMC, DIMC);
    addres_kernel<<<(ROWS * DIMC) / 256, blk, 0, stream>>>(x, bigC, proj_b + l * DIMC, DIMC);
    // ---- MLP ----
    ln_kernel<_Float16><<<ROWS, blk, 0, stream>>>(x, n2w + l * DIMC, n2b + l * DIMC, h16);
    cvt_kernel<<<(DIMC * MLPC) / 256, blk, 0, stream>>>(fc1_w + (size_t)l * DIMC * MLPC, w16);
    gemm_wmma<<<dim3(MLPC / BN, ROWS / BM), blk, 0, stream>>>(h16, w16, bigC, ROWS, MLPC, DIMC);
    gelu_kernel<<<(ROWS * MLPC) / 256, blk, 0, stream>>>(bigC, fc1_b + l * MLPC, m16);
    cvt_kernel<<<(MLPC * DIMC) / 256, blk, 0, stream>>>(fc2_w + (size_t)l * MLPC * DIMC, w16);
    gemm_wmma<<<dim3(DIMC / BN, ROWS / BM), blk, 0, stream>>>(m16, w16, bigC, ROWS, DIMC, MLPC);
    addres_kernel<<<(ROWS * DIMC) / 256, blk, 0, stream>>>(x, bigC, fc2_b + l * DIMC, DIMC);
  }

  ln_kernel<float><<<ROWS, blk, 0, stream>>>(x, fnw, fnb, bigC);
  fuse_kernel<<<(BB * TT * DIMC) / 256, blk, 0, stream>>>(bigC, out_fused);
}
